// CrossModalTFBlockV2_86440511799421
// MI455X (gfx1250) — matrix operand order repaired and checked by reading, not compiled, not run on hardware
//
#include <hip/hip_runtime.h>
#include <hip/hip_bf16.h>

// ---------------------------------------------------------------------------
// CrossModalTFBlockV2 for gfx1250 (MI455X). wave32, WMMA f32_16x16x32_f16,
// async global->LDS staging (ASYNCcnt) for all shared tiles.
// ---------------------------------------------------------------------------

typedef __attribute__((ext_vector_type(16))) _Float16 v16h;
typedef __attribute__((ext_vector_type(8)))  _Float16 v8h;
typedef __attribute__((ext_vector_type(8)))  float    v8f;

#define DEV static __device__ __forceinline__

DEV v8f wmma32(v16h a, v16h b, v8f c) {
  return __builtin_amdgcn_wmma_f32_16x16x32_f16(
      /*neg_a=*/false, a, /*neg_b=*/false, b,
      /*c_mod=*/(short)0, c, /*reuse_a=*/false, /*reuse_b=*/false);
}

// Async copy of 16 bytes global->LDS, tracked with ASYNCcnt (per-lane addrs).
DEV void async_cp16(const _Float16* g, _Float16* l) {
  unsigned lds = (unsigned)(uintptr_t)l;            // LDS aperture: addr[31:0]
  unsigned long long ga = (unsigned long long)(uintptr_t)g;
  asm volatile("global_load_async_to_lds_b128 %0, %1, off"
               :: "v"(lds), "v"(ga) : "memory");
}
#define ASYNC_WAIT(n) asm volatile("s_wait_asynccnt " #n ::: "memory")

// A fragment (16x32 f16) from row-major [M x K], leading dim ld (ISA 7.12.2).
DEV v16h load_a_frag(const _Float16* base, int ld) {
  int lane = threadIdx.x & 31;
  int m  = lane & 15;
  int hi = lane >> 4;
  const _Float16* p = base + (size_t)m * ld + hi * 16;
  v8h lo = *(const v8h*)p;
  v8h hh = *(const v8h*)(p + 8);
  v16h r;
#pragma unroll
  for (int i = 0; i < 8; ++i) { r[i] = lo[i]; r[8 + i] = hh[i]; }
  return r;
}

// B fragment (32x16 f16); storage = row-major [Ncols x K] (col of B per row).
DEV v16h load_b_frag(const _Float16* base, int ld) {
  int lane = threadIdx.x & 31;
  int n  = lane & 15;
  int hi = lane >> 4;
  return *(const v16h*)(base + (size_t)n * ld + hi * 16);
}

DEV float redmax16(float v) {
  v = fmaxf(v, __shfl_xor(v, 1, 32));
  v = fmaxf(v, __shfl_xor(v, 2, 32));
  v = fmaxf(v, __shfl_xor(v, 4, 32));
  v = fmaxf(v, __shfl_xor(v, 8, 32));
  return v;
}
DEV float redsum16(float v) {
  v += __shfl_xor(v, 1, 32);
  v += __shfl_xor(v, 2, 32);
  v += __shfl_xor(v, 4, 32);
  v += __shfl_xor(v, 8, 32);
  return v;
}

// ---------------------------------------------------------------------------
// 2x2 mean pool.
// ---------------------------------------------------------------------------
__global__ __launch_bounds__(256) void pool2_kernel(
    const float* __restrict__ in, float* __restrict__ outP,
    _Float16* __restrict__ outR) {
  int tid = blockIdx.x * 256 + threadIdx.x;
  int n = tid & 1023;
  int c = (tid >> 10) % 384;
  int b = tid / (1024 * 384);
  int y = n >> 5, x = n & 31;
  const float* p = in + ((size_t)(b * 384 + c) * 64 + 2 * y) * 64 + 2 * x;
  float v = 0.25f * (p[0] + p[1] + p[64] + p[65]);
  if (outP) outP[(size_t)(b * 384 + c) * 1024 + n] = v;
  outR[(size_t)(b * 1024 + n) * 384 + c] = (_Float16)v;
}

__global__ __launch_bounds__(256) void wcvt_kernel(
    const float* __restrict__ W, const float* __restrict__ s,
    _Float16* __restrict__ dst, int K) {
  int tid = blockIdx.x * 256 + threadIdx.x;
  dst[tid] = (_Float16)(W[tid] * s[tid / K]);
}

__global__ __launch_bounds__(256) void biascat_kernel(
    const float* __restrict__ a, const float* __restrict__ b,
    const float* __restrict__ c, float* __restrict__ dst) {
  int i = blockIdx.x * 256 + threadIdx.x;
  dst[i] = (i < 256) ? a[i] : ((i < 512) ? b[i - 256] : c[i - 512]);
}

// ---------------------------------------------------------------------------
// WMMA GEMM with async-LDS double-buffered staging.
// Y[M,Nout] = A[M,K](f16) @ W[Nout,K](f16)^T + bias, scatter epilogue modes.
// 256 threads = 8 waves (4 M x 2 N); WG tile 64x128, K-chunk 32.
// LDS chunk image: [0,2048) = A 64x32 ; [2048,6144) = W 128x32.
// ---------------------------------------------------------------------------
DEV void gemm_stage(const _Float16* __restrict__ A,
                    const _Float16* __restrict__ W, int K,
                    int row0blk, int col0blk, int kk, _Float16* dst) {
  int t = threadIdx.x;
#pragma unroll
  for (int i = 0; i < 3; ++i) {
    int u = t + i * 256;          // 768 x 16B units
    int eo = u * 8;
    const _Float16* src;
    if (eo < 2048) {
      src = A + (size_t)(row0blk + (eo >> 5)) * K + kk + (eo & 31);
    } else {
      int e2 = eo - 2048;
      src = W + (size_t)(col0blk + (e2 >> 5)) * K + kk + (e2 & 31);
    }
    async_cp16(src, dst + eo);
  }
}

__global__ __launch_bounds__(256) void gemm_wmma_kernel(
    const _Float16* __restrict__ A, int K,
    const _Float16* __restrict__ W, const float* __restrict__ bias,
    int M, int Nout, int mode,
    _Float16* __restrict__ p0, _Float16* __restrict__ p1,
    _Float16* __restrict__ p2,
    float* __restrict__ o32, const float* __restrict__ res) {
  __shared__ __align__(32) _Float16 sbuf[2][6144];   // 24KB double buffer
  int wid  = threadIdx.x >> 5;
  int lane = threadIdx.x & 31;
  int wm = wid & 3, wn = wid >> 2;
  int row0blk = blockIdx.x * 64;
  int col0blk = blockIdx.y * 128;
  int row0 = row0blk + wm * 16;
  int col0 = col0blk + wn * 64;
  (void)M; (void)Nout;

  v8f acc[4] = {};
  int nch = K >> 5;
  gemm_stage(A, W, K, row0blk, col0blk, 0, sbuf[0]);
  for (int c = 0; c < nch; ++c) {
    _Float16* cur = sbuf[c & 1];
    if (c + 1 < nch) {
      gemm_stage(A, W, K, row0blk, col0blk, (c + 1) * 32, sbuf[(c + 1) & 1]);
      ASYNC_WAIT(3);                 // current chunk's 3 issues complete
    } else {
      ASYNC_WAIT(0);
    }
    __syncthreads();
    v16h a = load_a_frag(cur + wm * 16 * 32, 32);
#pragma unroll
    for (int t = 0; t < 4; ++t) {
      v16h b = load_b_frag(cur + 2048 + (wn * 64 + t * 16) * 32, 32);
      acc[t] = wmma32(a, b, acc[t]);
    }
    __syncthreads();                 // all waves done reading before re-stage
  }

  int cl = lane & 15;
  int rowoff = (lane >> 4) * 8;
#pragma unroll
  for (int t = 0; t < 4; ++t) {
    int o = col0 + t * 16 + cl;
    float bs = bias[o];
#pragma unroll
    for (int i = 0; i < 8; ++i) {
      int row = row0 + rowoff + i;
      float val = acc[t][i] + bs;
      int bb = row >> 10;
      int n  = row & 1023;
      if (mode == 0) {                       // fused QKV scatter
        if (o < 512) {                       // q / kT: [b,h,n,32]
          int oo = o & 255;
          int hh = oo >> 5, kd = oo & 31;
          _Float16* dst = (o < 256) ? p0 : p1;
          dst[(size_t)((bb * 8 + hh) * 1024 + n) * 32 + kd] = (_Float16)val;
        } else {                             // vT: [b,h,d,n]
          int c2 = o - 512, hh = c2 >> 6, d = c2 & 63;
          p2[(size_t)((bb * 8 + hh) * 64 + d) * 1024 + n] = (_Float16)val;
        }
      } else if (mode == 1) {                // ekT: [b,h,n,32]
        int hh = o >> 5, kd = o & 31;
        p0[(size_t)((bb * 8 + hh) * 1024 + n) * 32 + kd] = (_Float16)val;
      } else if (mode == 2) {                // x = rgbP + Wp(relu(attn))
        size_t ci = (size_t)(bb * 384 + o) * 1024 + n;
        float xv = res[ci] + val;
        o32[ci] = xv;
        p0[(size_t)row * 384 + o] = (_Float16)xv;
      } else if (mode == 3) {                // h1: [b,1536,n] ch-major
        p0[(size_t)(bb * 1536 + o) * 1024 + n] = (_Float16)val;
      } else {                               // final: d_out = x + W2(h2)
        size_t ci = (size_t)(bb * 384 + o) * 1024 + n;
        o32[ci] = res[ci] + val;
      }
    }
  }
}

// ---------------------------------------------------------------------------
// Dual-softmax flash attention with async-LDS K/EK/V staging.
// 4 waves/block, all on the same (b,h): K/EK/V tiles staged once per block.
// LDS chunk image: [0,1024)=kT 32x32, [1024,2048)=ekT 32x32, [2048,4096)=vT 64x32.
// ---------------------------------------------------------------------------
DEV void attn_stage(const _Float16* __restrict__ kB,
                    const _Float16* __restrict__ eB,
                    const _Float16* __restrict__ vB, int jc, _Float16* dst) {
  int t = threadIdx.x;              // 0..127
#pragma unroll
  for (int i = 0; i < 4; ++i) {
    int u = t + i * 128;            // 512 x 16B units
    int eo = u * 8;
    const _Float16* src;
    if (eo < 1024) {
      src = kB + (size_t)(jc + (eo >> 5)) * 32 + (eo & 31);
    } else if (eo < 2048) {
      int e2 = eo - 1024;
      src = eB + (size_t)(jc + (e2 >> 5)) * 32 + (e2 & 31);
    } else {
      int e2 = eo - 2048;
      src = vB + (size_t)(e2 >> 5) * 1024 + jc + (e2 & 31);
    }
    async_cp16(src, dst + eo);
  }
}

__global__ __launch_bounds__(128) void attn_wmma_kernel(
    const _Float16* __restrict__ q,  const _Float16* __restrict__ kT,
    const _Float16* __restrict__ vT, const _Float16* __restrict__ ekT,
    _Float16* __restrict__ outO) {
  __shared__ __align__(32) _Float16 sb[2][4096];     // 16KB double buffer
  __shared__ __align__(32) _Float16 pb[4][16 * 32];  // per-wave P transpose
  int wid  = threadIdx.x >> 5;
  int lane = threadIdx.x & 31;
  int bh = blockIdx.x >> 4;                 // 4 m-tiles/block, 16 blocks per bh
  int mt = (blockIdx.x & 15) * 4 + wid;     // 0..63

  const _Float16* kB = kT + (size_t)bh * 1024 * 32;
  const _Float16* eB = ekT + (size_t)bh * 1024 * 32;
  const _Float16* vB = vT + (size_t)bh * 64 * 1024;

  v16h aq = load_a_frag(q + (size_t)(bh * 1024 + mt * 16) * 32, 32);

  v8f accr[4] = {}, accc[4] = {};
  float mr[8], sr[8], mc[8], sc[8], corr[8];
#pragma unroll
  for (int i = 0; i < 8; ++i) { mr[i] = mc[i] = -1e30f; sr[i] = sc[i] = 0.f; }
  int cl = lane & 15, rowoff = (lane >> 4) * 8;
  _Float16* P = pb[wid];

  attn_stage(kB, eB, vB, 0, sb[0]);
  for (int jc = 0; jc < 1024; jc += 32) {
    _Float16* cur = sb[(jc >> 5) & 1];
    if (jc + 32 < 1024) {
      attn_stage(kB, eB, vB, jc + 32, sb[((jc >> 5) + 1) & 1]);
      ASYNC_WAIT(4);
    } else {
      ASYNC_WAIT(0);
    }
    __syncthreads();

    v16h kb0 = load_b_frag(cur, 32);
    v16h kb1 = load_b_frag(cur + 16 * 32, 32);
    v16h eb0 = load_b_frag(cur + 1024, 32);
    v16h eb1 = load_b_frag(cur + 1024 + 16 * 32, 32);
    v8f z = {};
    v8f s0 = wmma32(aq, kb0, z);
    v8f s1 = wmma32(aq, kb1, z);
    v8f u0 = wmma32(aq, eb0, z);
    v8f u1 = wmma32(aq, eb1, z);

    v16h bv[4];
#pragma unroll
    for (int t = 0; t < 4; ++t)
      bv[t] = load_b_frag(cur + 2048 + t * 16 * 32, 32);

    // ---- rgb softmax path
#pragma unroll
    for (int i = 0; i < 8; ++i) {
      float mn = fmaxf(mr[i], redmax16(fmaxf(s0[i], s1[i])));
      corr[i] = __expf(mr[i] - mn);
      mr[i] = mn;
      float e0 = __expf(s0[i] - mn);
      float e1 = __expf(s1[i] - mn);
      sr[i] = sr[i] * corr[i] + redsum16(e0 + e1);
      P[(rowoff + i) * 32 + cl]      = (_Float16)e0;
      P[(rowoff + i) * 32 + 16 + cl] = (_Float16)e1;
    }
    v16h ap = load_a_frag(P, 32);
#pragma unroll
    for (int t = 0; t < 4; ++t) {
#pragma unroll
      for (int i = 0; i < 8; ++i) accr[t][i] *= corr[i];
      accr[t] = wmma32(ap, bv[t], accr[t]);
    }

    // ---- cross-modal softmax path
#pragma unroll
    for (int i = 0; i < 8; ++i) {
      float mn = fmaxf(mc[i], redmax16(fmaxf(u0[i], u1[i])));
      corr[i] = __expf(mc[i] - mn);
      mc[i] = mn;
      float e0 = __expf(u0[i] - mn);
      float e1 = __expf(u1[i] - mn);
      sc[i] = sc[i] * corr[i] + redsum16(e0 + e1);
      P[(rowoff + i) * 32 + cl]      = (_Float16)e0;
      P[(rowoff + i) * 32 + 16 + cl] = (_Float16)e1;
    }
    v16h ac = load_a_frag(P, 32);
#pragma unroll
    for (int t = 0; t < 4; ++t) {
#pragma unroll
      for (int i = 0; i < 8; ++i) accc[t][i] *= corr[i];
      accc[t] = wmma32(ac, bv[t], accc[t]);
    }
    __syncthreads();
  }

  int bb = bh >> 3, hh = bh & 7;
#pragma unroll
  for (int t = 0; t < 4; ++t) {
#pragma unroll
    for (int i = 0; i < 8; ++i) {
      int n = mt * 16 + rowoff + i;
      int d = t * 16 + cl;
      float val = accr[t][i] / sr[i] + 0.5f * accc[t][i] / sc[i];
      val = fmaxf(val, 0.f);                 // relu before Wp projection
      outO[(size_t)(bb * 1024 + n) * 512 + hh * 64 + d] = (_Float16)val;
    }
  }
}

// ---------------------------------------------------------------------------
// Depthwise 3x3 (SAME) + bias + relu.
// ---------------------------------------------------------------------------
__global__ __launch_bounds__(256) void dwconv_kernel(
    const _Float16* __restrict__ h1, const float* __restrict__ Wdw,
    const float* __restrict__ bdw, _Float16* __restrict__ h2) {
  int tid = blockIdx.x * 256 + threadIdx.x;
  int n = tid & 1023;
  int c = (tid >> 10) % 1536;
  int b = tid / (1024 * 1536);
  int y = n >> 5, x = n & 31;
  const _Float16* src = h1 + (size_t)(b * 1536 + c) * 1024;
  float acc = 0.f;
#pragma unroll
  for (int ky = 0; ky < 3; ++ky) {
    int yy = y + ky - 1;
    if (yy < 0 || yy > 31) continue;
#pragma unroll
    for (int kx = 0; kx < 3; ++kx) {
      int xx = x + kx - 1;
      if (xx < 0 || xx > 31) continue;
      acc += (float)src[yy * 32 + xx] * Wdw[c * 9 + ky * 3 + kx];
    }
  }
  acc = fmaxf(acc + bdw[c], 0.f);
  h2[(size_t)(b * 1024 + n) * 1536 + c] = (_Float16)acc;
}

// ---------------------------------------------------------------------------
extern "C" void kernel_launch(void* const* d_in, const int* in_sizes, int n_in,
                              void* d_out, int out_size, void* d_ws,
                              size_t ws_size, hipStream_t stream) {
  (void)in_sizes; (void)n_in; (void)out_size; (void)ws_size;
  const float* rgb_x  = (const float*)d_in[0];
  const float* edge_x = (const float*)d_in[1];
  const float* Wq = (const float*)d_in[2];
  const float* sq = (const float*)d_in[3];
  const float* bq = (const float*)d_in[4];
  const float* Wk = (const float*)d_in[5];
  const float* sk = (const float*)d_in[6];
  const float* bk = (const float*)d_in[7];
  const float* Wv = (const float*)d_in[8];
  const float* sv = (const float*)d_in[9];
  const float* bv = (const float*)d_in[10];
  const float* Wek = (const float*)d_in[11];
  const float* sek = (const float*)d_in[12];
  const float* bek = (const float*)d_in[13];
  const float* Wp = (const float*)d_in[14];
  const float* sp = (const float*)d_in[15];
  const float* bp = (const float*)d_in[16];
  const float* W1 = (const float*)d_in[17];
  const float* s1 = (const float*)d_in[18];
  const float* b1 = (const float*)d_in[19];
  const float* Wdw = (const float*)d_in[20];
  const float* bdw = (const float*)d_in[21];
  const float* W2 = (const float*)d_in[22];
  const float* s2 = (const float*)d_in[23];
  const float* b2 = (const float*)d_in[24];

  char* ws = (char*)d_ws;
  size_t off = 0;
#define WSALLOC(name, T, count)                                              \
  T* name = (T*)(ws + off);                                                  \
  off += (((size_t)(count) * sizeof(T)) + 255) & ~(size_t)255;

  WSALLOC(rgbP,  float,    8 * 384 * 1024)
  size_t q_off = off + 2 * ((8192 * 384 * 2 + 255) & ~(size_t)255);
  WSALLOC(Xr,    _Float16, 8192 * 384)
  WSALLOC(Xe,    _Float16, 8192 * 384)
  WSALLOC(qb,    _Float16, 64 * 1024 * 32)
  WSALLOC(kTb,   _Float16, 64 * 1024 * 32)
  WSALLOC(vTb,   _Float16, 64 * 64 * 1024)
  WSALLOC(ekTb,  _Float16, 64 * 1024 * 32)
  WSALLOC(attnO, _Float16, 8192 * 512)
  WSALLOC(x32,   float,    8 * 384 * 1024)
  WSALLOC(x16,   _Float16, 8192 * 384)
  WSALLOC(h2,    _Float16, 8192 * 1536)
  WSALLOC(Wqkvf, _Float16, 1024 * 384)
  WSALLOC(Wekf,  _Float16, 256 * 384)
  WSALLOC(Wpf,   _Float16, 384 * 512)
  WSALLOC(W1f,   _Float16, 1536 * 384)
  WSALLOC(W2f,   _Float16, 384 * 1536)
  WSALLOC(bqkv,  float,    1024)
  _Float16* h1 = (_Float16*)(ws + q_off);   // aliases dead q..attnO region

  pool2_kernel<<<12288, 256, 0, stream>>>(rgb_x, rgbP, Xr);
  pool2_kernel<<<12288, 256, 0, stream>>>(edge_x, nullptr, Xe);

  wcvt_kernel<<<(256 * 384) / 256, 256, 0, stream>>>(Wq, sq, Wqkvf, 384);
  wcvt_kernel<<<(256 * 384) / 256, 256, 0, stream>>>(Wk, sk, Wqkvf + 256 * 384, 384);
  wcvt_kernel<<<(512 * 384) / 256, 256, 0, stream>>>(Wv, sv, Wqkvf + 512 * 384, 384);
  wcvt_kernel<<<(256 * 384) / 256, 256, 0, stream>>>(Wek, sek, Wekf, 384);
  wcvt_kernel<<<(384 * 512) / 256, 256, 0, stream>>>(Wp, sp, Wpf, 512);
  wcvt_kernel<<<(1536 * 384) / 256, 256, 0, stream>>>(W1, s1, W1f, 384);
  wcvt_kernel<<<(384 * 1536) / 256, 256, 0, stream>>>(W2, s2, W2f, 1536);
  biascat_kernel<<<4, 256, 0, stream>>>(bq, bk, bv, bqkv);

  gemm_wmma_kernel<<<dim3(128, 8), 256, 0, stream>>>(
      Xr, 384, Wqkvf, bqkv, 8192, 1024, 0, qb, kTb, vTb, nullptr, nullptr);
  gemm_wmma_kernel<<<dim3(128, 2), 256, 0, stream>>>(
      Xe, 384, Wekf, bek, 8192, 256, 1, ekTb, nullptr, nullptr, nullptr, nullptr);

  attn_wmma_kernel<<<1024, 128, 0, stream>>>(qb, kTb, vTb, ekTb, attnO);

  gemm_wmma_kernel<<<dim3(128, 3), 256, 0, stream>>>(
      attnO, 512, Wpf, bp, 8192, 384, 2, x16, nullptr, nullptr, x32, rgbP);

  gemm_wmma_kernel<<<dim3(128, 12), 256, 0, stream>>>(
      x16, 384, W1f, b1, 8192, 1536, 3, h1, nullptr, nullptr, nullptr, nullptr);

  dwconv_kernel<<<49152, 256, 0, stream>>>(h1, Wdw, bdw, h2);

  gemm_wmma_kernel<<<dim3(128, 3), 256, 0, stream>>>(
      h2, 1536, W2f, b2, 8192, 384, 4, nullptr, nullptr, nullptr,
      (float*)d_out, x32);
}
